// FlashInfer_KeyFormer_317827580190
// MI455X (gfx1250) — compile-verified, matrix operand order checked by reading, not run on hardware
//
#include <hip/hip_runtime.h>

// ---------------------------------------------------------------------------
// MI455X / gfx1250 fused attention + keyformer pipeline (wave32, WMMA bf16)
// ---------------------------------------------------------------------------

typedef __attribute__((ext_vector_type(16))) __bf16 v16bf;
typedef __attribute__((ext_vector_type(8)))  float  v8f;

static constexpr int kM = 1024;
static constexpr int kN = 1024;
static constexpr int kD = 64;
static constexpr int kP = 4096;
static constexpr int kH = 16;
static constexpr int kL = kP + kM;   // 5120
static constexpr int kK = 1024;      // projection reduction dim

// ---- workspace layout (bytes) ----
static constexpr size_t WS_XB   = 0;                                   // X   bf16 [1024][1024]
static constexpr size_t WS_WT   = WS_XB  + 2097152;                    // Wt  bf16 3x[1024][1024] (W^T)
static constexpr size_t WS_QB   = WS_WT  + 3u * 2097152;               // Q   bf16 [1024][1024]
static constexpr size_t WS_KC   = WS_QB  + 2097152;                    // Kc  bf16 [H][L][D]
static constexpr size_t WS_VT   = WS_KC  + (size_t)kH * kL * kD * 2;   // Vt  bf16 [H][D][L]
static constexpr size_t WS_SMAX = WS_VT  + (size_t)kH * kL * kD * 2;   // f32 [H*M]
static constexpr size_t WS_SSUM = WS_SMAX + (size_t)kH * kM * 4;
static constexpr size_t WS_PMAX = WS_SSUM + (size_t)kH * kM * 4;
static constexpr size_t WS_PSUM = WS_PMAX + (size_t)kH * kM * 4;

// float -> bf16 round-to-nearest-even
static __device__ __forceinline__ unsigned short f2bf(float f) {
  unsigned int u = __float_as_uint(f);
  u += 0x7FFFu + ((u >> 16) & 1u);
  return (unsigned short)(u >> 16);
}

union FragU {
  v16bf v;
  unsigned int u[8];
};

static __device__ __forceinline__ v8f vzero() {
  v8f z = {0.f, 0.f, 0.f, 0.f, 0.f, 0.f, 0.f, 0.f};
  return z;
}

// Load a 16x32 bf16 fragment in the ISA 16-bit A layout:
//   lane (lr in 0..15, half in 0..1) holds row lr; VGPR v holds K pair
//   k0 = (v&3)*2 + (v>>2)*16 + half*8.  All K pairs are contiguous 4B loads.
// For B we load B^T rows with the same pattern (lr = output column).
static __device__ __forceinline__ v16bf load_frag(const unsigned short* base,
                                                  int strideUS, int lr, int half) {
  FragU f;
#pragma unroll
  for (int v = 0; v < 8; ++v) {
    int k0 = ((v & 3) << 1) + ((v >> 2) << 4) + (half << 3);
    f.u[v] = *reinterpret_cast<const unsigned int*>(base + (size_t)lr * strideUS + k0);
  }
  return f.v;
}

static __device__ __forceinline__ v8f wmma_bf16(v16bf a, v16bf b, v8f c) {
  return __builtin_amdgcn_wmma_f32_16x16x32_bf16(false, a, false, b,
                                                 (short)0, c, false, false);
}

// ---------------------------------------------------------------------------
// prep kernels
// ---------------------------------------------------------------------------
__global__ void k_cvt_bf16(const float* __restrict__ s,
                           unsigned short* __restrict__ d, int n) {
  int i = blockIdx.x * blockDim.x + threadIdx.x;
  if (i < n) d[i] = f2bf(s[i]);
}

// cache_K [H][P][D] f32 -> Kc [H][L][D] bf16 (rows 0..P-1)
__global__ void k_cvt_cacheK(const float* __restrict__ s,
                             unsigned short* __restrict__ d) {
  int i = blockIdx.x * blockDim.x + threadIdx.x;
  if (i < kH * kP * kD) {
    int h = i >> 18;               // P*D = 262144 = 2^18
    int rem = i & 262143;
    d[(size_t)h * (kL * kD) + rem] = f2bf(s[i]);
  }
}

// tiled transpose + convert: dst[c*dstStride + r] = bf16(src[r*cols + c])
__global__ void k_transpose_cvt(const float* __restrict__ src,
                                unsigned short* __restrict__ dst,
                                int rows, int cols, int dstStride,
                                long srcBatch, long dstBatch) {
  __shared__ float tile[32][33];
  const float* s = src + (size_t)blockIdx.z * srcBatch;
  unsigned short* d = dst + (size_t)blockIdx.z * dstBatch;
  int c0 = blockIdx.x * 32, r0 = blockIdx.y * 32;
#pragma unroll
  for (int j = threadIdx.y; j < 32; j += 8) {
    int r = r0 + j, c = c0 + threadIdx.x;
    if (r < rows && c < cols) tile[j][threadIdx.x] = s[(size_t)r * cols + c];
  }
  __syncthreads();
#pragma unroll
  for (int j = threadIdx.y; j < 32; j += 8) {
    int c = c0 + j, r = r0 + threadIdx.x;
    if (r < rows && c < cols)
      d[(size_t)c * dstStride + r] = f2bf(tile[threadIdx.x][j]);
  }
}

// ---------------------------------------------------------------------------
// QKV projections: bf16 WMMA GEMM, each wave owns a 16(m) x 64(n) tile.
// mat 0 -> Qb[m][n]; mat 1 -> Kc[h][P+m][d]; mat 2 -> Vt[h][d][P+m]
// ---------------------------------------------------------------------------
__global__ __launch_bounds__(128) void k_gemm_qkv(
    const unsigned short* __restrict__ Xb, const unsigned short* __restrict__ Wt,
    unsigned short* __restrict__ Qb, unsigned short* __restrict__ Kc,
    unsigned short* __restrict__ Vt) {
  int lane = threadIdx.x & 31, lr = lane & 15, half = lane >> 4;
  int wid = (blockIdx.x << 2) + (threadIdx.x >> 5);   // 3072 waves total
  int mat = wid >> 10;
  int rem = wid & 1023;
  int m0 = (rem >> 4) << 4;   // 64 row tiles
  int n0 = (rem & 15) << 6;   // 16 col tiles of 64

  v8f acc[4] = {vzero(), vzero(), vzero(), vzero()};
  const unsigned short* A = Xb + (size_t)m0 * kK;
  const unsigned short* B = Wt + (size_t)mat * (size_t)kK * kN;

  for (int kc = 0; kc < kK; kc += 32) {
    v16bf a = load_frag(A + kc, kK, lr, half);
#pragma unroll
    for (int j = 0; j < 4; ++j) {
      v16bf b = load_frag(B + (size_t)(n0 + j * 16) * kK + kc, kK, lr, half);
      acc[j] = wmma_bf16(a, b, acc[j]);
    }
  }

#pragma unroll
  for (int i = 0; i < 8; ++i) {
    int m = m0 + i + (half << 3);
#pragma unroll
    for (int j = 0; j < 4; ++j) {
      int n = n0 + (j << 4) + lr;
      unsigned short bv = f2bf(acc[j][i]);
      if (mat == 0) {
        Qb[(size_t)m * kN + n] = bv;
      } else {
        int h = n >> 6, dd = n & 63;
        if (mat == 1)
          Kc[(size_t)h * (kL * kD) + (size_t)(kP + m) * kD + dd] = bv;
        else
          Vt[(size_t)h * (kD * kL) + (size_t)dd * kL + (kP + m)] = bv;
      }
    }
  }
}

// ---------------------------------------------------------------------------
// Pass 1: per-row online softmax stats (plain + perturbed), one wave per
// (h, 16-row m tile).  Per-lane online (max,sum); one butterfly merge at end.
// ---------------------------------------------------------------------------
__global__ __launch_bounds__(32) void k_stats(
    const unsigned short* __restrict__ Qb, const unsigned short* __restrict__ Kc,
    const float* __restrict__ noise, float* __restrict__ smax,
    float* __restrict__ ssum, float* __restrict__ pmax,
    float* __restrict__ psum) {
  int lane = threadIdx.x, lr = lane & 15, half = lane >> 4;
  int h = blockIdx.x >> 6;
  int m0 = (blockIdx.x & 63) << 4;

  const unsigned short* qb = Qb + (size_t)m0 * kN + h * kD;
  v16bf a0 = load_frag(qb, kN, lr, half);
  v16bf a1 = load_frag(qb + 32, kN, lr, half);
  const unsigned short* kcb = Kc + (size_t)h * (kL * kD);
  const float* nb = noise + ((size_t)h * kM + m0) * kL;

  float mW[8], sW[8], mP[8], sP[8];
#pragma unroll
  for (int i = 0; i < 8; ++i) {
    mW[i] = -3.0e38f; sW[i] = 0.f; mP[i] = -3.0e38f; sP[i] = 0.f;
  }

  for (int l0 = 0; l0 < kL; l0 += 16) {
    const unsigned short* kb = kcb + (size_t)l0 * kD;
    __builtin_prefetch(kb + 16 * kD, 0, 1);
    v16bf b0 = load_frag(kb, kD, lr, half);
    v16bf b1 = load_frag(kb + 32, kD, lr, half);
    v8f c = vzero();
    c = wmma_bf16(a0, b0, c);
    c = wmma_bf16(a1, b1, c);
#pragma unroll
    for (int i = 0; i < 8; ++i) {
      int row = i + (half << 3);
      float x = c[i];
      float nm = fmaxf(mW[i], x);
      sW[i] = sW[i] * __expf(mW[i] - nm) + __expf(x - nm);
      mW[i] = nm;
      float p = (x + nb[(size_t)row * kL + l0 + lr]) * (1.0f / 1.5f);
      nm = fmaxf(mP[i], p);
      sP[i] = sP[i] * __expf(mP[i] - nm) + __expf(p - nm);
      mP[i] = nm;
    }
  }

  // merge (max,sum) pairs across the 16-lane half (masks stay inside half)
#pragma unroll
  for (int i = 0; i < 8; ++i) {
#pragma unroll
    for (int msk = 1; msk <= 8; msk <<= 1) {
      float om = __shfl_xor(mW[i], msk, 32);
      float os = __shfl_xor(sW[i], msk, 32);
      float nm = fmaxf(mW[i], om);
      sW[i] = sW[i] * __expf(mW[i] - nm) + os * __expf(om - nm);
      mW[i] = nm;
      om = __shfl_xor(mP[i], msk, 32);
      os = __shfl_xor(sP[i], msk, 32);
      nm = fmaxf(mP[i], om);
      sP[i] = sP[i] * __expf(mP[i] - nm) + os * __expf(om - nm);
      mP[i] = nm;
    }
  }
  if (lr == 0) {
#pragma unroll
    for (int i = 0; i < 8; ++i) {
      int idx = h * kM + m0 + i + (half << 3);
      smax[idx] = mW[i]; ssum[idx] = sW[i];
      pmax[idx] = mP[i]; psum[idx] = sP[i];
    }
  }
}

// ---------------------------------------------------------------------------
// Pass 2: recompute scores, emit weights + perturb_out exactly once, and fuse
// the weights @ V_cat GEMM (weights tile -> LDS -> A-fragment -> WMMA).
// ---------------------------------------------------------------------------
__global__ __launch_bounds__(32) void k_finalize(
    const unsigned short* __restrict__ Qb, const unsigned short* __restrict__ Kc,
    const unsigned short* __restrict__ Vt, const float* __restrict__ noise,
    const float* __restrict__ smax, const float* __restrict__ ssum,
    const float* __restrict__ pmax, const float* __restrict__ psum,
    float* __restrict__ outO, float* __restrict__ outW,
    float* __restrict__ outP) {
  __shared__ unsigned short lw[16 * 32];   // 16(m) x 32(l) bf16 weights tile
  int lane = threadIdx.x, lr = lane & 15, half = lane >> 4;
  int h = blockIdx.x >> 6;
  int m0 = (blockIdx.x & 63) << 4;

  const unsigned short* qb = Qb + (size_t)m0 * kN + h * kD;
  v16bf a0 = load_frag(qb, kN, lr, half);
  v16bf a1 = load_frag(qb + 32, kN, lr, half);
  const unsigned short* kcb = Kc + (size_t)h * (kL * kD);
  const unsigned short* vtb = Vt + (size_t)h * (kD * kL);
  const float* nb = noise + ((size_t)h * kM + m0) * kL;
  float* wb = outW + ((size_t)h * kM + m0) * kL;
  float* pb = outP + ((size_t)h * kM + m0) * kL;

  float sm[8], si[8], pm[8], pi[8];
#pragma unroll
  for (int i = 0; i < 8; ++i) {
    int idx = h * kM + m0 + i + (half << 3);
    sm[i] = smax[idx]; si[i] = 1.0f / ssum[idx];
    pm[i] = pmax[idx]; pi[i] = 1.0f / psum[idx];
  }

  v8f o[4] = {vzero(), vzero(), vzero(), vzero()};

  for (int l0 = 0; l0 < kL; l0 += 32) {
    v8f c[2];
#pragma unroll
    for (int t = 0; t < 2; ++t) {
      const unsigned short* kb = kcb + (size_t)(l0 + t * 16) * kD;
      __builtin_prefetch(kb + 32 * kD, 0, 1);
      v16bf b0 = load_frag(kb, kD, lr, half);
      v16bf b1 = load_frag(kb + 32, kD, lr, half);
      v8f z = vzero();
      z = wmma_bf16(a0, b0, z);
      z = wmma_bf16(a1, b1, z);
      c[t] = z;
    }
#pragma unroll
    for (int t = 0; t < 2; ++t) {
#pragma unroll
      for (int i = 0; i < 8; ++i) {
        int row = i + (half << 3);
        size_t col = (size_t)(l0 + t * 16 + lr);
        float x = c[t][i];
        float wv = __expf(x - sm[i]) * si[i];
        wb[(size_t)row * kL + col] = wv;
        float pv = __expf((x + nb[(size_t)row * kL + col]) * (1.0f / 1.5f) - pm[i]) * pi[i];
        pb[(size_t)row * kL + col] = pv;
        lw[row * 32 + t * 16 + lr] = f2bf(wv);
      }
    }
    // LDS is in-order per wave; make sure stores land before fragment reads
    asm volatile("s_wait_dscnt 0x0" ::: "memory");
    FragU fa;
#pragma unroll
    for (int v = 0; v < 8; ++v) {
      int k0 = ((v & 3) << 1) + ((v >> 2) << 4) + (half << 3);
      fa.u[v] = *reinterpret_cast<const unsigned int*>(&lw[lr * 32 + k0]);
    }
#pragma unroll
    for (int j = 0; j < 4; ++j) {
      v16bf b = load_frag(vtb + (size_t)(j * 16) * kL + l0, kL, lr, half);
      o[j] = wmma_bf16(fa.v, b, o[j]);
    }
  }

#pragma unroll
  for (int i = 0; i < 8; ++i) {
    int m = m0 + i + (half << 3);
#pragma unroll
    for (int j = 0; j < 4; ++j)
      outO[(size_t)m * kN + h * kD + j * 16 + lr] = o[j][i];
  }
}

// ---------------------------------------------------------------------------
extern "C" void kernel_launch(void* const* d_in, const int* in_sizes, int n_in,
                              void* d_out, int out_size, void* d_ws, size_t ws_size,
                              hipStream_t stream) {
  (void)in_sizes; (void)n_in; (void)out_size; (void)ws_size;
  const float* X      = (const float*)d_in[0];
  const float* Wq     = (const float*)d_in[1];
  const float* Wk     = (const float*)d_in[2];
  const float* Wv     = (const float*)d_in[3];
  const float* cacheK = (const float*)d_in[4];
  const float* cacheV = (const float*)d_in[5];
  const float* noise  = (const float*)d_in[6];

  char* ws = (char*)d_ws;
  unsigned short* Xb = (unsigned short*)(ws + WS_XB);
  unsigned short* Wt = (unsigned short*)(ws + WS_WT);
  unsigned short* Qb = (unsigned short*)(ws + WS_QB);
  unsigned short* Kc = (unsigned short*)(ws + WS_KC);
  unsigned short* Vt = (unsigned short*)(ws + WS_VT);
  float* smax = (float*)(ws + WS_SMAX);
  float* ssum = (float*)(ws + WS_SSUM);
  float* pmax = (float*)(ws + WS_PMAX);
  float* psum = (float*)(ws + WS_PSUM);

  float* outO = (float*)d_out;
  float* outW = outO + (size_t)kM * kN;
  float* outP = outW + (size_t)kH * kM * kL;

  // prep: bf16 conversions / transposes
  k_cvt_bf16<<<4096, 256, 0, stream>>>(X, Xb, kM * kK);
  k_transpose_cvt<<<dim3(32, 32, 1), dim3(32, 8), 0, stream>>>(
      Wq, Wt + 0 * (size_t)kK * kN, kK, kN, kK, 0, 0);
  k_transpose_cvt<<<dim3(32, 32, 1), dim3(32, 8), 0, stream>>>(
      Wk, Wt + 1 * (size_t)kK * kN, kK, kN, kK, 0, 0);
  k_transpose_cvt<<<dim3(32, 32, 1), dim3(32, 8), 0, stream>>>(
      Wv, Wt + 2 * (size_t)kK * kN, kK, kN, kK, 0, 0);
  k_cvt_cacheK<<<16384, 256, 0, stream>>>(cacheK, Kc);
  k_transpose_cvt<<<dim3(2, 128, 16), dim3(32, 8), 0, stream>>>(
      cacheV, Vt, kP, kD, kL, (long)kP * kD, (long)kD * kL);

  // QKV projections (WMMA bf16)
  k_gemm_qkv<<<768, 128, 0, stream>>>(Xb, Wt, Qb, Kc, Vt);

  // pass 1: softmax stats for both plain and perturbed paths
  k_stats<<<kH * (kM / 16), 32, 0, stream>>>(Qb, Kc, noise, smax, ssum, pmax, psum);

  // pass 2: emit weights/perturb, fused weights @ V_cat
  k_finalize<<<kH * (kM / 16), 32, 0, stream>>>(Qb, Kc, Vt, noise, smax, ssum,
                                                pmax, psum, outO, outW, outP);
}
